// VariableSelectionNetwork_19937238188299
// MI455X (gfx1250) — matrix operand order repaired and checked by reading, not compiled, not run on hardware
//
#include <hip/hip_runtime.h>

typedef __attribute__((ext_vector_type(16))) _Float16 v16h;
typedef __attribute__((ext_vector_type(8)))  float    v8f;

#define F_DIM          64
#define H_DIM          64
#define K_DIM          4096        // F*H
#define POS_TOTAL      16384       // 32*512
#define POS_PER_BLOCK  64
#define THREADS        256         // 8 wave32 waves: 4 M-tiles x 2 K-halves

// ---------------------------------------------------------------------------
// Pre-pass: pack Ws [4096 x 64] fp32 (k-major) into f16 WMMA B-fragments.
// Fragment layout (CDNA5 ISA 7.12.2, 16-bit B 32x16):
//   lane = grp*16 + n16 ; half j 0..7 -> k = ks*32 + grp*8 + j ; j 8..15 -> k = ks*32 + 16 + grp*8 + (j-8)
// Packed: wsPack[ ((ks*32 + lane)*4 + nt)*16 + j ]  => 128 contiguous B/lane/slice
// ---------------------------------------------------------------------------
__global__ __launch_bounds__(256) void pack_ws_kernel(const float* __restrict__ Ws,
                                                      _Float16* __restrict__ wsPack) {
    unsigned u = blockIdx.x * blockDim.x + threadIdx.x;     // 131072 threads, 2 halves each
    if (u >= 131072u) return;
    int j    = (int)(u & 7u) * 2;
    int nt   = (int)(u >> 3) & 3;
    int lane = (int)(u >> 5) & 31;
    int ks   = (int)(u >> 10);
    int n = nt * 16 + (lane & 15);
    int k = ks * 32 + (j & 8) * 2 + (lane >> 4) * 8 + (j & 7);
    _Float16 h0 = (_Float16)Ws[(size_t)k * 64 + n];
    _Float16 h1 = (_Float16)Ws[(size_t)(k + 1) * 64 + n];
    size_t out = ((size_t)(ks * 32 + lane) * 4 + nt) * 16 + j;
    wsPack[out]     = h0;
    wsPack[out + 1] = h1;
}

__device__ __forceinline__ float elu_f(float z) {
    return z > 0.0f ? z : (__expf(z) - 1.0f);
}

// ---------------------------------------------------------------------------
// Main fused kernel, 64 positions / block, 8 waves:
//   wave = (khalf, mtile):  mtile owns 16 positions, khalf owns 2048 of K=4096
//   phase 1: partial logits via v_wmma_f32_16x16x32_f16 (64 k-slices/wave)
//   phase 1b: K-split reduction in LDS (in place, race-free)
//   phase 2: softmax in LDS
//   phase 3: selected = weights . elu(x*Wf+bf), 8 positions/wave
// ---------------------------------------------------------------------------
__global__ __launch_bounds__(THREADS) void vsn_kernel(
    const float* __restrict__ x,     // [16384, 64]
    const float* __restrict__ Wf,    // [64, 64]
    const float* __restrict__ bf,    // [64, 64]
    const float* __restrict__ bs,    // [64]
    const _Float16* __restrict__ wsPack,
    float* __restrict__ outSel,      // [16384, 64]
    float* __restrict__ outW)        // [16384, 64]
{
    __shared__ float s_x [K_DIM];    // transposed: [f][pos_local]
    __shared__ float s_wf[K_DIM];    // [f][h]
    __shared__ float s_bf[K_DIM];    // [f][h]
    __shared__ float s_lw[K_DIM];    // logits -> weights, [pos_local][64]

    const int tid      = threadIdx.x;
    const int posBlock = blockIdx.x * POS_PER_BLOCK;

    // ---- cooperative stage-in (b128 loads) -------------------------------
    for (int i = tid * 4; i < K_DIM; i += THREADS * 4) {
        int p = i >> 6;          // pos_local
        int f0 = i & 63;
        float4 xv = *(const float4*)&x[(size_t)(posBlock + p) * 64 + f0];
        s_x[(f0 + 0) * 64 + p] = xv.x;
        s_x[(f0 + 1) * 64 + p] = xv.y;
        s_x[(f0 + 2) * 64 + p] = xv.z;
        s_x[(f0 + 3) * 64 + p] = xv.w;
        *(float4*)&s_wf[i] = *(const float4*)&Wf[i];
        *(float4*)&s_bf[i] = *(const float4*)&bf[i];
    }
    __syncthreads();

    const int lane  = tid & 31;
    const int wv    = tid >> 5;      // 0..7
    const int mtile = wv & 3;        // 16-position M tile
    const int khalf = wv >> 2;       // K split half
    const int grp   = lane >> 4;
    const int mloc  = lane & 15;

    // ---- phase 1: WMMA over this wave's half of K ------------------------
    v8f acc0 = {}, acc1 = {}, acc2 = {}, acc3 = {};
    const int ksBeg = khalf * 64;                       // wave-uniform
    const _Float16* bPtr = wsPack + (size_t)lane * 64 + (size_t)ksBeg * 2048;

    // structurally-uniform trip count (64) -> scalar loop, EXEC untouched
    for (int i = 0; i < 64; ++i) {
        const int   ks = ksBeg + i;
        const int   f  = ks >> 1;
        const int   h0 = (ks & 1) << 5;
        const float xv = s_x[f * 64 + mtile * 16 + mloc];

        // A-fragment (16-bit A 16x32, ISA 7.12.2):
        //  j 0..7  -> klocal = grp*8 + j ; j 8..15 -> klocal = 16 + grp*8 + (j-8)
        const float* wrow = &s_wf[f * 64 + h0 + grp * 8];
        const float* brow = &s_bf[f * 64 + h0 + grp * 8];
        v16h a;
#pragma unroll
        for (int j = 0; j < 8; ++j)
            a[j] = (_Float16)elu_f(fmaf(xv, wrow[j], brow[j]));
#pragma unroll
        for (int j = 0; j < 8; ++j)
            a[8 + j] = (_Float16)elu_f(fmaf(xv, wrow[16 + j], brow[16 + j]));

        const v16h* bp = (const v16h*)bPtr;
        v16h b0 = bp[0], b1 = bp[1], b2 = bp[2], b3 = bp[3];
        bPtr += 2048;                                   // next k-slice (32*64 halves)

        acc0 = __builtin_amdgcn_wmma_f32_16x16x32_f16(false, a, false, b0, (short)0, acc0, false, false);
        acc1 = __builtin_amdgcn_wmma_f32_16x16x32_f16(false, a, false, b1, (short)0, acc1, false, false);
        acc2 = __builtin_amdgcn_wmma_f32_16x16x32_f16(false, a, false, b2, (short)0, acc2, false, false);
        acc3 = __builtin_amdgcn_wmma_f32_16x16x32_f16(false, a, false, b3, (short)0, acc3, false, false);
    }

    // ---- phase 1b: K-split reduction in LDS ------------------------------
    // C layout: row M = r + 8*grp, col N = mloc (+ 16*nt)
    if (khalf == 0) {
        const float bs0 = bs[mloc], bs1 = bs[16 + mloc], bs2 = bs[32 + mloc], bs3 = bs[48 + mloc];
#pragma unroll
        for (int r = 0; r < 8; ++r) {
            int row = mtile * 16 + grp * 8 + r;
            s_lw[row * 64 + mloc]      = acc0[r] + bs0;
            s_lw[row * 64 + 16 + mloc] = acc1[r] + bs1;
            s_lw[row * 64 + 32 + mloc] = acc2[r] + bs2;
            s_lw[row * 64 + 48 + mloc] = acc3[r] + bs3;
        }
    }
    __syncthreads();
    if (khalf == 1) {
        // each (row,col) owned by exactly one lane of one khalf-1 wave: race-free RMW
#pragma unroll
        for (int r = 0; r < 8; ++r) {
            int row = mtile * 16 + grp * 8 + r;
            s_lw[row * 64 + mloc]      += acc0[r];
            s_lw[row * 64 + 16 + mloc] += acc1[r];
            s_lw[row * 64 + 32 + mloc] += acc2[r];
            s_lw[row * 64 + 48 + mloc] += acc3[r];
        }
    }
    __syncthreads();

    // ---- phase 2: softmax over 64 features, in place ---------------------
    if (tid < POS_PER_BLOCK) {
        float* row = &s_lw[tid * 64];
        float mx = row[0];
#pragma unroll
        for (int n = 1; n < 64; ++n) mx = fmaxf(mx, row[n]);
        float sum = 0.0f;
#pragma unroll
        for (int n = 0; n < 64; ++n) { float e = __expf(row[n] - mx); row[n] = e; sum += e; }
        float inv = 1.0f / sum;
#pragma unroll
        for (int n = 0; n < 64; ++n) row[n] *= inv;
    }
    __syncthreads();

    // coalesced weights write-out
    for (int i = tid * 4; i < K_DIM; i += THREADS * 4)
        *(float4*)&outW[(size_t)posBlock * 64 + i] = *(const float4*)&s_lw[i];

    // ---- phase 3: selected[h] = sum_f w[f] * elu(x*Wf+bf), 8 pos/wave ----
    const int hh = lane * 2;                     // each lane: h, h+1
    for (int m = 0; m < 8; ++m) {
        const int pl = wv * 8 + m;
        float a0 = 0.0f, a1 = 0.0f;
#pragma unroll 4
        for (int f = 0; f < 64; ++f) {
            float  w   = s_lw[pl * 64 + f];                    // broadcast
            float  xv  = s_x [f * 64 + pl];                    // broadcast
            float2 wfp = *(const float2*)&s_wf[f * 64 + hh];   // ds_load_b64
            float2 bfp = *(const float2*)&s_bf[f * 64 + hh];
            float z0 = elu_f(fmaf(xv, wfp.x, bfp.x));
            float z1 = elu_f(fmaf(xv, wfp.y, bfp.y));
            a0 = fmaf(w, z0, a0);
            a1 = fmaf(w, z1, a1);
        }
        float2 o; o.x = a0; o.y = a1;
        *(float2*)&outSel[(size_t)(posBlock + pl) * 64 + hh] = o;   // global_store_b64
    }
}

// ---------------------------------------------------------------------------
extern "C" void kernel_launch(void* const* d_in, const int* in_sizes, int n_in,
                              void* d_out, int out_size, void* d_ws, size_t ws_size,
                              hipStream_t stream) {
    const float* x  = (const float*)d_in[0];   // [32,512,64]
    const float* Wf = (const float*)d_in[1];   // [64,64]
    const float* bf = (const float*)d_in[2];   // [64,64]
    const float* Ws = (const float*)d_in[3];   // [4096,64]
    const float* bs = (const float*)d_in[4];   // [64]

    float* out    = (float*)d_out;
    float* outSel = out;                                   // [16384,64]
    float* outW   = out + (size_t)POS_TOTAL * H_DIM;       // [16384,64]

    _Float16* wsPack = (_Float16*)d_ws;                    // needs 512 KB

    pack_ws_kernel<<<512, 256, 0, stream>>>(Ws, wsPack);
    vsn_kernel<<<POS_TOTAL / POS_PER_BLOCK, THREADS, 0, stream>>>(
        x, Wf, bf, bs, wsPack, outSel, outW);
}
